// RCNNHead_12618613916199
// MI455X (gfx1250) — matrix operand order repaired and checked by reading, not compiled, not run on hardware
//
#include <hip/hip_runtime.h>
#include <stdint.h>

typedef __attribute__((ext_vector_type(16))) _Float16 v16h;
typedef __attribute__((ext_vector_type(8)))  float    v8f;
typedef int v4i __attribute__((vector_size(16)));   // pointee type of async-LDS builtins

#define NROIS  2048
#define CROP   7
#define P2     49
#define CCH    256
#define HH     200
#define WW     200
#define FAN_IN (CCH * P2)   // 12544
#define HID    1024

union FragH {
    v16h     h;
    uint2    u2[4];
    unsigned u[8];
};

// ---------------------------------------------------------------------------
// CDNA5 async global<->LDS copy (ASYNCcnt path), with safe fallback.
// Builtin signature (from hipcc diagnostics): (v4i AS1*, v4i AS3*, imm, imm).
// ---------------------------------------------------------------------------
#if defined(__has_builtin)
#if __has_builtin(__builtin_amdgcn_global_load_async_to_lds_b128) && \
    __has_builtin(__builtin_amdgcn_global_store_async_from_lds_b128) && \
    __has_builtin(__builtin_amdgcn_s_wait_asynccnt)
#define USE_ASYNC_LDS 1
#endif
#endif
#ifndef USE_ASYNC_LDS
#define USE_ASYNC_LDS 0
#endif

#define AS1 __attribute__((address_space(1)))
#define AS3 __attribute__((address_space(3)))

// generic shared pointer -> LDS offset (ISA: generic maps to LDS by dropping
// upper bits), reinterpreted as a 32-bit AS(3) pointer.
__device__ __forceinline__ AS3 v4i* lds_cast(const void* p) {
    return (AS3 v4i*)(unsigned)(uintptr_t)p;
}
__device__ __forceinline__ AS1 v4i* gbl_cast(const void* p) {
    return (AS1 v4i*)(uintptr_t)p;
}

// 16B global -> LDS copy (async DMA when available)
__device__ __forceinline__ void copy16_g2l(const _Float16* g, unsigned* l) {
#if USE_ASYNC_LDS
    __builtin_amdgcn_global_load_async_to_lds_b128(gbl_cast(g), lds_cast(l), 0, 0);
#else
    uint4 t = *(const uint4*)g;
    ((uint2*)l)[0] = make_uint2(t.x, t.y);
    ((uint2*)l)[1] = make_uint2(t.z, t.w);
#endif
}
// 16B LDS -> global copy (async DMA when available)
__device__ __forceinline__ void copy16_l2g(_Float16* g, const _Float16* l) {
#if USE_ASYNC_LDS
    __builtin_amdgcn_global_store_async_from_lds_b128(gbl_cast(g), lds_cast(l), 0, 0);
#else
    *(uint4*)g = *(const uint4*)l;
#endif
}
// wave-local async completion + workgroup visibility
__device__ __forceinline__ void async_join() {
#if USE_ASYNC_LDS
    __builtin_amdgcn_s_wait_asynccnt(0);
#endif
    __syncthreads();
}

// ---------------------------------------------------------------------------
// fp32 -> f16 weight conversion (once per launch; deterministic)
// ---------------------------------------------------------------------------
__global__ void cvt_f32_to_f16(const float* __restrict__ in,
                               _Float16* __restrict__ out, int n) {
    int i = blockIdx.x * blockDim.x + threadIdx.x;
    int stride = gridDim.x * blockDim.x;
    for (; i < n; i += stride) out[i] = (_Float16)in[i];
}

// ---------------------------------------------------------------------------
// RoIAlign: one block per ROI, thread = channel, 49 bins each.
// x is L2-resident (164MB < 192MB L2). Stage row in LDS, stream it out with
// async LDS->global B128 DMA (S_ENDPGM drains ASYNCcnt before kernel end).
// Output layout matches reference: crops[r, c*49 + p], p = py*7 + px.
// ---------------------------------------------------------------------------
__global__ __launch_bounds__(256) void roi_align_kernel(
    const float* __restrict__ x, const float* __restrict__ rois,
    _Float16* __restrict__ crops)
{
    __shared__ __align__(16) _Float16 tile[FAN_IN];
    const int r = blockIdx.x;
    const int c = threadIdx.x;

    const float bi  = rois[r * 5 + 0];
    const float rx1 = rois[r * 5 + 1];
    const float ry1 = rois[r * 5 + 2];
    const float rx2 = rois[r * 5 + 3];
    const float ry2 = rois[r * 5 + 4];
    const int   b   = (int)bi;
    const float bw  = (rx2 - rx1) * (1.0f / CROP);
    const float bh  = (ry2 - ry1) * (1.0f / CROP);

    const float* xb = x + (size_t)(b * CCH + c) * (HH * WW);

    for (int p = 0; p < P2; ++p) {
        const int pyi = p / CROP;
        const int pxi = p % CROP;
        float sx = rx1 + ((float)pxi + 0.5f) * bw;
        float sy = ry1 + ((float)pyi + 0.5f) * bh;
        sx = fminf(fmaxf(sx, 0.0f), (float)(WW - 1));
        sy = fminf(fmaxf(sy, 0.0f), (float)(HH - 1));
        const float fx0 = floorf(sx), fy0 = floorf(sy);
        const int x0 = (int)fx0, y0 = (int)fy0;
        const int x1 = min(x0 + 1, WW - 1);
        const int y1 = min(y0 + 1, HH - 1);
        const float wx = sx - fx0, wy = sy - fy0;

        const float v00 = xb[y0 * WW + x0];
        const float v01 = xb[y0 * WW + x1];
        const float v10 = xb[y1 * WW + x0];
        const float v11 = xb[y1 * WW + x1];

        const float val = v00 * (1.0f - wy) * (1.0f - wx)
                        + v01 * (1.0f - wy) * wx
                        + v10 * wy * (1.0f - wx)
                        + v11 * wy * wx;
        tile[c * P2 + p] = (_Float16)val;
    }
    __syncthreads();
    // 12544 halves = 1568 x 16B chunks, coalesced LDS -> global
    _Float16* orow = crops + (size_t)r * FAN_IN;
    for (int ch = threadIdx.x; ch < FAN_IN / 8; ch += 256)
        copy16_l2g(orow + ch * 8, tile + ch * 8);
}

// ---------------------------------------------------------------------------
// GEMM: Out[M,N] = relu?(A[M,K] * Wt[N,K]^T + bias[N]), f16 in, f32 acc, f16 out.
// Block tile 128x64, BK=64, 8 waves (4x2), wave tile 32x32 (2x2 WMMA tiles).
// Double-buffered LDS filled by async global->LDS B128 DMA; padded row stride
// (64+8 halves = 144B = 9x16B) keeps frag reads 16B-aligned and bank-spread.
// Requires M%128==0, N%64==0, K%64==0 (true for all calls here).
// ---------------------------------------------------------------------------
#define BM 128
#define BN 64
#define BK 64
#define LDS_STRIDE_U 36   // (64+8) halves / 2 uints

__global__ __launch_bounds__(256) void gemm_bias_act_f16(
    const _Float16* __restrict__ A,
    const _Float16* __restrict__ Wt,
    const float* __restrict__ bias,
    _Float16* __restrict__ Out,
    int M, int N, int K, int relu)
{
    __shared__ __align__(16) unsigned As[2][BM * LDS_STRIDE_U];
    __shared__ __align__(16) unsigned Bs[2][BN * LDS_STRIDE_U];

    const int tid  = threadIdx.x;
    const int lane = tid & 31;
    const int wave = tid >> 5;
    const int wm   = wave & 3;     // 4 waves along M
    const int wn   = wave >> 2;    // 2 waves along N
    const int hi   = lane >> 4;    // half-wave select
    const int ln   = lane & 15;

    const int gm0 = blockIdx.y * BM;
    const int gn0 = blockIdx.x * BN;
    const int S   = K / BK;

    auto issue_stage = [&](int s, int nb) {
        const int ks = s * BK;
        #pragma unroll
        for (int i = 0; i < 4; ++i) {
            const int chunk = tid + 256 * i;           // 1024 chunks of 8 halves
            const int row = chunk >> 3, c8 = chunk & 7;
            copy16_g2l(A + (size_t)(gm0 + row) * K + ks + c8 * 8,
                       &As[nb][row * LDS_STRIDE_U + c8 * 4]);
        }
        #pragma unroll
        for (int i = 0; i < 2; ++i) {
            const int chunk = tid + 256 * i;           // 512 chunks
            const int row = chunk >> 3, c8 = chunk & 7;
            copy16_g2l(Wt + (size_t)(gn0 + row) * K + ks + c8 * 8,
                       &Bs[nb][row * LDS_STRIDE_U + c8 * 4]);
        }
    };

    v8f acc[2][2] = {};

    issue_stage(0, 0);
    async_join();

    for (int s = 0; s < S; ++s) {
        const int nb = s & 1;
        if (s + 1 < S) issue_stage(s + 1, nb ^ 1);   // DMA overlaps WMMAs below

        #pragma unroll
        for (int kk = 0; kk < BK; kk += 32) {
            FragH af[2], bf[2];
            // A frag: lane row = ln; 16 contiguous halves at kk+16*hi
            #pragma unroll
            for (int ti = 0; ti < 2; ++ti) {
                const int rowu = (wm * 32 + ti * 16 + ln) * LDS_STRIDE_U;
                const unsigned* sp = &As[nb][0];
                const int u0 = rowu + (kk >> 1) + hi * 8;
                const int u1 = rowu + (kk >> 1) + hi * 8 + 4;
                af[ti].u2[0] = *(const uint2*)&sp[u0];
                af[ti].u2[1] = *(const uint2*)&sp[u0 + 2];
                af[ti].u2[2] = *(const uint2*)&sp[u1];
                af[ti].u2[3] = *(const uint2*)&sp[u1 + 2];
            }
            // B frag: lane row (=N col) = ln; 16 contiguous halves at kk+16*hi
            #pragma unroll
            for (int tj = 0; tj < 2; ++tj) {
                const int rowu = (wn * 32 + tj * 16 + ln) * LDS_STRIDE_U;
                const unsigned* sp = &Bs[nb][0];
                const int u0 = rowu + (kk >> 1) + hi * 8;
                #pragma unroll
                for (int j = 0; j < 4; ++j)
                    bf[tj].u2[j] = *(const uint2*)&sp[u0 + 2 * j];
            }
            #pragma unroll
            for (int ti = 0; ti < 2; ++ti)
                #pragma unroll
                for (int tj = 0; tj < 2; ++tj)
                    acc[ti][tj] = __builtin_amdgcn_wmma_f32_16x16x32_f16(
                        false, af[ti].h, false, bf[tj].h,
                        (short)0, acc[ti][tj], false, false);
        }

        if (s + 1 < S) async_join();
    }

    // Epilogue: C/D layout -> lanes 0-15: M=v, N=ln; lanes 16-31: M=v+8
    #pragma unroll
    for (int ti = 0; ti < 2; ++ti) {
        #pragma unroll
        for (int tj = 0; tj < 2; ++tj) {
            const int gn = gn0 + wn * 32 + tj * 16 + ln;
            const float bv = bias[gn];
            #pragma unroll
            for (int v = 0; v < 8; ++v) {
                const int gm = gm0 + wm * 32 + ti * 16 + v + hi * 8;
                float r = acc[ti][tj][v] + bv;
                if (relu) r = fmaxf(r, 0.0f);
                Out[(size_t)gm * N + gn] = (_Float16)r;
            }
        }
    }
}

// ---------------------------------------------------------------------------
// Heads: logits [2048,2] and bbox [2048,8] fused into one 16-col WMMA tile.
// One wave per 16 ROIs; cols 0-1 = wc rows, 2-9 = wb rows, 10-15 masked zero.
// fp32 weights converted to f16 on the fly; output fp32 to d_out.
// ---------------------------------------------------------------------------
__global__ __launch_bounds__(256) void head_kernel(
    const _Float16* __restrict__ h2,
    const float* __restrict__ wc, const float* __restrict__ bc,
    const float* __restrict__ wb, const float* __restrict__ bb,
    float* __restrict__ logits, float* __restrict__ bbox)
{
    const int gw   = (blockIdx.x * blockDim.x + threadIdx.x) >> 5;  // 0..127
    const int lane = threadIdx.x & 31;
    const int n    = lane & 15;
    const int hi   = lane >> 4;
    const int m0   = gw * 16;

    const float* wrow = wc;
    float msk = 0.0f;
    if (n < 2)       { wrow = wc + n * HID;       msk = 1.0f; }
    else if (n < 10) { wrow = wb + (n - 2) * HID; msk = 1.0f; }

    const _Float16* arow = h2 + (size_t)(m0 + n) * HID;

    v8f acc = {};
    for (int k = 0; k < HID; k += 32) {
        FragH a, b;
        const uint4 a0 = *(const uint4*)(arow + k + hi * 16);
        const uint4 a1 = *(const uint4*)(arow + k + hi * 16 + 8);
        a.u[0] = a0.x; a.u[1] = a0.y; a.u[2] = a0.z; a.u[3] = a0.w;
        a.u[4] = a1.x; a.u[5] = a1.y; a.u[6] = a1.z; a.u[7] = a1.w;
        const float* wp = wrow + k + hi * 16;
        #pragma unroll
        for (int j = 0; j < 16; ++j) b.h[j] = (_Float16)(msk * wp[j]);
        acc = __builtin_amdgcn_wmma_f32_16x16x32_f16(
            false, a.h, false, b.h, (short)0, acc, false, false);
    }

    #pragma unroll
    for (int v = 0; v < 8; ++v) {
        const int m = m0 + v + hi * 8;
        const float r = acc[v];
        if (n < 2)       logits[m * 2 + n]     = r + bc[n];
        else if (n < 10) bbox[m * 8 + (n - 2)] = r + bb[n - 2];
    }
}

// ---------------------------------------------------------------------------
// Launch
// ---------------------------------------------------------------------------
extern "C" void kernel_launch(void* const* d_in, const int* in_sizes, int n_in,
                              void* d_out, int out_size, void* d_ws, size_t ws_size,
                              hipStream_t stream) {
    (void)in_sizes; (void)n_in; (void)out_size; (void)ws_size;
    const float* x    = (const float*)d_in[0];
    const float* rois = (const float*)d_in[1];
    const float* w1   = (const float*)d_in[2];
    const float* b1   = (const float*)d_in[3];
    const float* w2   = (const float*)d_in[4];
    const float* b2   = (const float*)d_in[5];
    const float* wc   = (const float*)d_in[6];
    const float* bc   = (const float*)d_in[7];
    const float* wb   = (const float*)d_in[8];
    const float* bb   = (const float*)d_in[9];
    float* out = (float*)d_out;

    char* ws = (char*)d_ws;
    size_t off = 0;
    auto alloc = [&](size_t bytes) {
        char* p = ws + off;
        off += (bytes + 255) & ~(size_t)255;
        return p;
    };
    _Float16* crops = (_Float16*)alloc((size_t)NROIS * FAN_IN * 2);  // 51.4 MB
    _Float16* w1h   = (_Float16*)alloc((size_t)HID * FAN_IN * 2);    // 25.7 MB
    _Float16* w2h   = (_Float16*)alloc((size_t)HID * HID * 2);       //  2.1 MB
    _Float16* h1    = (_Float16*)alloc((size_t)NROIS * HID * 2);     //  4.2 MB
    _Float16* h2    = (_Float16*)alloc((size_t)NROIS * HID * 2);     //  4.2 MB

    cvt_f32_to_f16<<<2048, 256, 0, stream>>>(w1, w1h, HID * FAN_IN);
    cvt_f32_to_f16<<<512, 256, 0, stream>>>(w2, w2h, HID * HID);

    roi_align_kernel<<<NROIS, 256, 0, stream>>>(x, rois, crops);

    gemm_bias_act_f16<<<dim3(HID / BN, NROIS / BM), 256, 0, stream>>>(
        crops, w1h, b1, h1, NROIS, HID, FAN_IN, 1);
    gemm_bias_act_f16<<<dim3(HID / BN, NROIS / BM), 256, 0, stream>>>(
        h1, w2h, b2, h2, NROIS, HID, HID, 1);

    head_kernel<<<16, 256, 0, stream>>>(h2, wc, bc, wb, bb, out, out + NROIS * 2);
}
